// GCNNetTianshouPPO_CRITIC_44976897524021
// MI455X (gfx1250) — matrix operand order repaired and checked by reading, not compiled, not run on hardware
//
#include <hip/hip_runtime.h>
#include <stdint.h>

// ---------------------------------------------------------------------------
// GCN network forward for MI455X (gfx1250, wave32, WMMA).
//   conv: t = h @ W (bf16 WMMA, f32 acc, fragment-packed weights) ;
//   agg in LDS (ds_add_f32) with fused self-loop init + bias + leaky-relu.
//   head: algebraically folded  out[b] = (sum_n h2) . (Wl2@Ws) + N*(bl2.Ws) + bs
// ---------------------------------------------------------------------------

typedef __attribute__((ext_vector_type(16))) __bf16 v16bf;
typedef __attribute__((ext_vector_type(8)))  float  v8f;

union Frag16 { v16bf v; uint16_t u[16]; uint32_t w[8]; uint4 q[2]; };

static constexpr int Bq   = 32;
static constexpr int Nq   = 2000;
static constexpr int Eq   = 32000;
static constexpr int CIN  = 128;
static constexpr int Hq   = 256;
static constexpr int OUTq = 128;
static constexpr int NODES = Bq * Nq;          // 64000
#define LEAKY 0.01f

__device__ __forceinline__ float bf2f(uint16_t u) {
    union { uint32_t i; float f; } c; c.i = ((uint32_t)u) << 16; return c.f;
}
__device__ __forceinline__ uint16_t f2bf(float f) {
    union { float f; uint32_t i; } c; c.f = f;
    uint32_t r = (c.i + 0x7FFFu + ((c.i >> 16) & 1u)) >> 16;   // round-nearest-even
    return (uint16_t)r;
}

// ------------------------------- utilities ---------------------------------
__global__ void fill_f32(float* p, float v, int n) {
    int i = blockIdx.x * blockDim.x + threadIdx.x;
    if (i < n) p[i] = v;
}
__global__ void cvt_f32_bf16(const float* __restrict__ in, uint16_t* __restrict__ out, int n) {
    int i = blockIdx.x * blockDim.x + threadIdx.x;
    if (i < n) out[i] = f2bf(in[i]);
}
// deg[dst] += 1 over all real edges (self-loop +1 folded into dinv kernel)
__global__ void deg_count(const int* __restrict__ links, float* __restrict__ deg) {
    int i = blockIdx.x * blockDim.x + threadIdx.x;
    if (i >= Bq * Eq) return;
    int b = i / Eq, e = i - b * Eq;
    int d = links[(size_t)b * 2 * Eq + Eq + e];
    atomicAdd(&deg[(size_t)b * Nq + d], 1.0f);
}
__global__ void dinv_k(const float* __restrict__ deg, float* __restrict__ dinv, int n) {
    int i = blockIdx.x * blockDim.x + threadIdx.x;
    if (i < n) dinv[i] = rsqrtf(deg[i] + 1.0f);   // +1 self loop; deg>=1 always
}

// ------------------- weight pack: f32 [K x 256] -> fragment order ----------
// Wp layout: [kb (K/32)][tile (16)][lane (32)][j (16)] bf16, so a wave's
// 32x16 B fragment is two contiguous b128 loads per lane.
// Fragment element map (lane: half=lane>>4, nl=lane&15):
//   u[2v]   = W[kb*32 + 2v   + half*16][tile*16 + nl]
//   u[2v+1] = W[kb*32 + 2v+1 + half*16][tile*16 + nl]
__global__ void pack_w_f32(const float* __restrict__ Wf, uint16_t* __restrict__ Wp, int K) {
    int idx = blockIdx.x * blockDim.x + threadIdx.x;   // ((kb*16+tile)*32+lane)*16+j
    if (idx >= K * 256) return;
    int j    = idx & 15;
    int lane = (idx >> 4) & 31;
    int tile = (idx >> 9) & 15;
    int kb   = idx >> 13;
    int half = lane >> 4, nl = lane & 15;
    int srcK = kb * 32 + 2 * (j >> 1) + (j & 1) + half * 16;
    int srcC = tile * 16 + nl;
    Wp[idx] = f2bf(Wf[(size_t)srcK * 256 + srcC]);
}

// ------------------------- bf16 WMMA GEMM (N=256) --------------------------
// Out[M x 256] (bf16) = A[M x K] (bf16) @ Wp (packed, f32 accum).
// Block = 8 waves; block computes 16 rows x 256 cols; A strip staged in LDS.
__global__ __launch_bounds__(256) void gemm_bf16_n256(
    const uint16_t* __restrict__ A, const uint16_t* __restrict__ Wp,
    uint16_t* __restrict__ Out, int K,
    const float* __restrict__ bias, int act)
{
    constexpr int Nout = 256;
    __shared__ __align__(16) uint16_t As[16 * 256];

    const int m0  = blockIdx.x * 16;
    const int tid = threadIdx.x;

    // stage A strip (contiguous 16*K bf16) into LDS
    {
        const uint4* src = (const uint4*)(A + (size_t)m0 * K);
        uint4* dst = (uint4*)As;
        const int n4 = (16 * K) / 8;
        for (int i = tid; i < n4; i += 256) dst[i] = src[i];
    }
    __syncthreads();

    const int wave = tid >> 5;
    const int lane = tid & 31;
    const int half = lane >> 4;
    const int m    = lane & 15;   // A row within tile
    const int nl   = lane & 15;   // C/D column within tile
    const int c0   = wave * 32;   // this wave owns cols [c0, c0+32)
    const int t0   = c0 >> 4;     // first of two 16-col tiles

    const uint4* Wq = (const uint4*)Wp;   // 2 uint4 per (kb,tile,lane)

    v8f acc0 = {0.f,0.f,0.f,0.f,0.f,0.f,0.f,0.f};
    v8f acc1 = {0.f,0.f,0.f,0.f,0.f,0.f,0.f,0.f};

    for (int kk = 0; kk < K; kk += 32) {
        const int kb = kk >> 5;
        // A fragment (16x32 bf16): two contiguous 16B runs in LDS
        Frag16 a;
        #pragma unroll
        for (int v = 0; v < 8; ++v) {
            int kp = 2 * v + half * 16;
            a.w[v] = *(const uint32_t*)&As[m * K + kk + kp];
        }
        // B fragments: contiguous b128 pairs from packed weights
        Frag16 b0, b1;
        {
            size_t base0 = ((size_t)(kb * 16 + t0)     * 32 + lane) * 2;
            size_t base1 = ((size_t)(kb * 16 + t0 + 1) * 32 + lane) * 2;
            b0.q[0] = Wq[base0];     b0.q[1] = Wq[base0 + 1];
            b1.q[0] = Wq[base1];     b1.q[1] = Wq[base1 + 1];
        }
        acc0 = __builtin_amdgcn_wmma_f32_16x16x32_bf16(false, a.v, false, b0.v,
                                                       (short)0, acc0, false, false);
        acc1 = __builtin_amdgcn_wmma_f32_16x16x32_bf16(false, a.v, false, b1.v,
                                                       (short)0, acc1, false, false);
    }

    // epilogue: bias values hoisted (depend only on column), leaky-relu, bf16
    float bias0 = 0.f, bias1 = 0.f;
    if (bias) { bias0 = bias[c0 + nl]; bias1 = bias[c0 + 16 + nl]; }
    #pragma unroll
    for (int r = 0; r < 8; ++r) {
        int row = m0 + r + half * 8;
        float v0 = acc0[r] + bias0;
        float v1 = acc1[r] + bias1;
        if (act) {
            v0 = v0 > 0.f ? v0 : LEAKY * v0;
            v1 = v1 > 0.f ? v1 : LEAKY * v1;
        }
        Out[(size_t)row * Nout + c0 + nl]      = f2bf(v0);
        Out[(size_t)row * Nout + c0 + 16 + nl] = f2bf(v1);
    }
}

// ----------------------- LDS-resident GCN aggregation ----------------------
// grid = (8 feature chunks, 32 graphs); block = 256; dyn LDS = 2000*32 floats
// (250 KB of the WGP's 320 KB). lds init = self-loop term; edges scatter with
// ds_add_f32; writeback fuses bias + leaky-relu + bf16 conversion.
__global__ __launch_bounds__(256) void agg_gcn(
    const uint16_t* __restrict__ T, const int* __restrict__ links,
    const float* __restrict__ dinv, const float* __restrict__ bias,
    uint16_t* __restrict__ Hout)
{
    extern __shared__ float lds[];                 // [Nq * 32]
    const int chunk = blockIdx.x;                  // 0..7
    const int b     = blockIdx.y;                  // 0..31
    const int cbase = chunk * 32;
    const int tid   = threadIdx.x;
    const size_t nodeBase = (size_t)b * Nq;

    // init with self-loop contribution: t[n] * dinv[n]^2
    for (int i = tid; i < Nq * 32; i += 256) {
        int n = i >> 5, f = i & 31;
        size_t g = nodeBase + n;
        float di = dinv[g];
        lds[i] = bf2f(T[g * 256 + cbase + f]) * di * di;
    }
    __syncthreads();

    // edge scatter: one wave per edge, lane = feature within chunk
    const int wave = tid >> 5;
    const int lane = tid & 31;
    const int* es = links + (size_t)b * 2 * Eq;
    const int* ed = es + Eq;
    for (int e = wave; e < Eq; e += 8) {
        int s = es[e];
        int d = ed[e];
        float nrm = dinv[nodeBase + s] * dinv[nodeBase + d];
        float msg = bf2f(T[(nodeBase + s) * 256 + cbase + lane]) * nrm;
        atomicAdd(&lds[d * 32 + lane], msg);       // ds_add_f32
    }
    __syncthreads();

    // writeback: bias + leaky-relu, bf16
    for (int i = tid; i < Nq * 32; i += 256) {
        int n = i >> 5, f = i & 31;
        float vv = lds[i] + bias[cbase + f];
        vv = vv > 0.f ? vv : LEAKY * vv;
        Hout[(nodeBase + n) * 256 + cbase + f] = f2bf(vv);
    }
}

// -------------------------------- head -------------------------------------
// S[b, f] = sum_n T[b, n, f]   (f32 accum from bf16)
__global__ __launch_bounds__(256) void graph_sum(const uint16_t* __restrict__ T,
                                                 float* __restrict__ S)
{
    int b = blockIdx.x, f = threadIdx.x;
    const uint16_t* p = T + (size_t)b * Nq * 256 + f;
    float acc = 0.f;
    for (int n = 0; n < Nq; ++n) acc += bf2f(p[(size_t)n * 256]);
    S[b * 256 + f] = acc;
}

// V = Wl2 @ Ws (256-vec);  C = Nq * (bl2 . Ws) + bs
__global__ __launch_bounds__(256) void fold_head(
    const float* __restrict__ Wl2, const float* __restrict__ bl2,
    const float* __restrict__ Ws,  const float* __restrict__ bs,
    float* __restrict__ V, float* __restrict__ C)
{
    __shared__ float red[128];
    int t = threadIdx.x;                           // 0..255
    float a = 0.f;
    for (int j = 0; j < OUTq; ++j) a += Wl2[t * OUTq + j] * Ws[j];
    V[t] = a;
    if (t < 128) red[t] = bl2[t] * Ws[t];
    __syncthreads();
    for (int s = 64; s > 0; s >>= 1) {
        if (t < s) red[t] += red[t + s];
        __syncthreads();
    }
    if (t == 0) C[0] = (float)Nq * red[0] + bs[0];
}

// out[b] = S[b,:] . V + C
__global__ __launch_bounds__(256) void head_out(
    const float* __restrict__ S, const float* __restrict__ V,
    const float* __restrict__ C, float* __restrict__ out)
{
    __shared__ float red[256];
    int b = blockIdx.x, t = threadIdx.x;
    red[t] = S[b * 256 + t] * V[t];
    __syncthreads();
    for (int s = 128; s > 0; s >>= 1) {
        if (t < s) red[t] += red[t + s];
        __syncthreads();
    }
    if (t == 0) out[b] = red[0] + C[0];
}

// ------------------------------ launch --------------------------------------
extern "C" void kernel_launch(void* const* d_in, const int* in_sizes, int n_in,
                              void* d_out, int out_size, void* d_ws, size_t ws_size,
                              hipStream_t stream)
{
    (void)in_sizes; (void)n_in; (void)out_size; (void)ws_size;

    const float* x   = (const float*)d_in[0];   // [B,N,CIN]
    const int*   el  = (const int*)  d_in[1];   // [B,2,E]
    const float* W1  = (const float*)d_in[2];
    const float* b1  = (const float*)d_in[3];
    const float* Wc  = (const float*)d_in[4];   // [3,H,H]
    const float* bc  = (const float*)d_in[5];   // [3,H]
    const float* Wl1 = (const float*)d_in[6];
    const float* bl1 = (const float*)d_in[7];
    const float* Wl2 = (const float*)d_in[8];
    const float* bl2 = (const float*)d_in[9];
    const float* Ws  = (const float*)d_in[10];
    const float* bs  = (const float*)d_in[11];
    float* out = (float*)d_out;

    char* w = (char*)d_ws;
    size_t off = 0;
    auto take = [&](size_t bytes) -> char* {
        char* p = w + off;
        off = (off + bytes + 255) & ~(size_t)255;
        return p;
    };
    float*    deg  = (float*)   take((size_t)NODES * 4);
    float*    dinv = (float*)   take((size_t)NODES * 4);
    uint16_t* w1p  = (uint16_t*)take((size_t)CIN * Hq * 2);      // packed
    uint16_t* wcp  = (uint16_t*)take((size_t)3 * Hq * Hq * 2);   // packed
    uint16_t* wl1p = (uint16_t*)take((size_t)Hq * Hq * 2);       // packed
    uint16_t* xb   = (uint16_t*)take((size_t)NODES * CIN * 2);
    uint16_t* hb   = (uint16_t*)take((size_t)NODES * Hq * 2);
    uint16_t* tb   = (uint16_t*)take((size_t)NODES * Hq * 2);
    float*    S    = (float*)   take((size_t)Bq * Hq * 4);
    float*    V    = (float*)   take((size_t)Hq * 4);
    float*    C    = (float*)   take(256);

    // degrees + symmetric normalization
    fill_f32 <<<(NODES + 255) / 256, 256, 0, stream>>>(deg, 0.f, NODES);
    deg_count<<<(Bq * Eq + 255) / 256, 256, 0, stream>>>(el, deg);
    dinv_k   <<<(NODES + 255) / 256, 256, 0, stream>>>(deg, dinv, NODES);

    // input conversion + fused weight convert/pack (fragment-native order)
    cvt_f32_bf16<<<((size_t)NODES * CIN + 255) / 256, 256, 0, stream>>>(x, xb, NODES * CIN);
    pack_w_f32<<<(CIN * Hq + 255) / 256, 256, 0, stream>>>(W1, w1p, CIN);
    for (int i = 0; i < 3; ++i)
        pack_w_f32<<<(Hq * Hq + 255) / 256, 256, 0, stream>>>(
            Wc + (size_t)i * Hq * Hq, wcp + (size_t)i * Hq * Hq, Hq);
    pack_w_f32<<<(Hq * Hq + 255) / 256, 256, 0, stream>>>(Wl1, wl1p, Hq);

    const size_t aggLds = (size_t)Nq * 32 * sizeof(float);   // 250 KB of 320 KB/WGP

    // conv1: x @ W1 -> t ; aggregate -> h (bias b1 + leaky fused)
    gemm_bf16_n256<<<NODES / 16, 256, 0, stream>>>(xb, w1p, tb, CIN, nullptr, 0);
    agg_gcn<<<dim3(8, Bq), 256, aggLds, stream>>>(tb, el, dinv, b1, hb);

    // conv2..4
    for (int i = 0; i < 3; ++i) {
        gemm_bf16_n256<<<NODES / 16, 256, 0, stream>>>(hb, wcp + (size_t)i * Hq * Hq,
                                                       tb, Hq, nullptr, 0);
        agg_gcn<<<dim3(8, Bq), 256, aggLds, stream>>>(tb, el, dinv, bc + i * Hq, hb);
    }

    // mlp1 with fused bias + leaky-relu
    gemm_bf16_n256<<<NODES / 16, 256, 0, stream>>>(hb, wl1p, tb, Hq, bl1, 1);

    // folded head
    graph_sum<<<Bq, 256, 0, stream>>>(tb, S);
    fold_head<<<1, 256, 0, stream>>>(Wl2, bl2, Ws, bs, V, C);
    head_out <<<Bq, 256, 0, stream>>>(S, V, C, out);
}
